// GAT_11948599017538
// MI455X (gfx1250) — compile-verified, matrix operand order checked by reading, not run on hardware
//
#include <hip/hip_runtime.h>
#include <hip/hip_bf16.h>
#include <math.h>

// ---------------------------------------------------------------------------
// GAT 2-layer forward for gfx1250 (MI455X). wave32.
//  - GEMMs: f16 WMMA (v_wmma_f32_16x16x32_f16), W-slab staged in LDS in
//    fragment-ready layout (one ds_load_b128 pair per lane per K-step).
//  - Edge passes: memory-bound atomic segment max / sum / weighted scatter,
//    with global_prefetch on gathered feature rows.
// ---------------------------------------------------------------------------

#define NN     50000        // nodes
#define EE     800000       // raw edges
#define ETOT   850000       // edges + self loops
#define INC    128          // input channels
#define HDC    128          // heads*hid = 8*16
#define HEADS  8
#define HID    16
#define OUTC   40
#define NEG    0.2f

typedef __attribute__((ext_vector_type(16))) _Float16 v16h;
typedef __attribute__((ext_vector_type(8)))  float    v8f;

// ---- order-preserving float<->int map for atomicMax-based segment max ------
__device__ __forceinline__ int   fenc(float f) { int b = __float_as_int(f); return b >= 0 ? b : (b ^ 0x7fffffff); }
__device__ __forceinline__ float fdec(int b)   { return __int_as_float(b >= 0 ? b : (b ^ 0x7fffffff)); }
__device__ __forceinline__ float lrelu(float v) { return v >= 0.0f ? v : NEG * v; }

// WMMA 16-bit fragment K mapping (ISA §7.12.2): element e of a lane's v16h,
// with lane-half `half`, maps to krel = (e<8 ? half*8+e : 16+half*8+(e-8)).
__device__ __forceinline__ int krel_of(int e, int half)
{
    return (e < 8) ? (half * 8 + e) : (16 + half * 8 + (e - 8));
}

// Load this lane's A fragment (16 f32 -> f16) for rows of X, K block kb.
__device__ __forceinline__ v16h load_a_frag(const float* __restrict__ X,
                                            int mrow, int kb, int half)
{
    const float* ap = X + (size_t)mrow * INC + kb + half * 8;
    float4 c0 = *(const float4*)(ap);
    float4 c1 = *(const float4*)(ap + 4);
    float4 c2 = *(const float4*)(ap + 16);
    float4 c3 = *(const float4*)(ap + 20);
    v16h a;
    a[0]=(_Float16)c0.x; a[1]=(_Float16)c0.y; a[2]=(_Float16)c0.z; a[3]=(_Float16)c0.w;
    a[4]=(_Float16)c1.x; a[5]=(_Float16)c1.y; a[6]=(_Float16)c1.z; a[7]=(_Float16)c1.w;
    a[8]=(_Float16)c2.x; a[9]=(_Float16)c2.y; a[10]=(_Float16)c2.z; a[11]=(_Float16)c2.w;
    a[12]=(_Float16)c3.x; a[13]=(_Float16)c3.y; a[14]=(_Float16)c3.z; a[15]=(_Float16)c3.w;
    return a;
}

// ===========================================================================
// GEMM1: H[NN,128] = X[NN,128] @ W1[128,128]   (f16 WMMA, f32 accumulate)
// Block = 8 waves; blockIdx.y = 16-col slab. W slab staged once per block
// into LDS in fragment-ready layout: bsh[kt*512 + lane*16 + e].
// ===========================================================================
__global__ void gat_gemm1_wmma(const float* __restrict__ X,
                               const float* __restrict__ W,
                               float* __restrict__ H)
{
    __shared__ __align__(32) _Float16 bsh[4 * 32 * 16];   // 4 K-steps, 4 KB
    const int tid = threadIdx.x;
    const int n0  = blockIdx.y << 4;

    // ---- cooperative stage: 2048 fragment slots / 256 threads = 8 each ----
#pragma unroll
    for (int j = 0; j < 8; ++j) {
        int t      = tid * 8 + j;          // slot = kt*512 + lane*16 + e
        int e      = t & 15;
        int lane_s = (t >> 4) & 31;
        int kt     = t >> 9;
        int k      = kt * 32 + krel_of(e, lane_s >> 4);
        bsh[t] = (_Float16)W[(size_t)k * HDC + (n0 + (lane_s & 15))];
    }
    __syncthreads();

    const int wave  = tid >> 5;
    const int lane  = tid & 31;
    const int mtile = blockIdx.x * 8 + wave;
    if (mtile >= (NN / 16)) return;               // wave-uniform: EXEC stays full
    const int half = lane >> 4;
    const int mrow = (mtile << 4) + (lane & 15);
    const int n    = n0 + (lane & 15);

    v8f acc = {};
#pragma unroll
    for (int kt = 0; kt < 4; ++kt) {
        v16h a = load_a_frag(X, mrow, kt * 32, half);
        v16h b = *(const v16h*)&bsh[kt * 512 + lane * 16];   // ds_load 32B
        acc = __builtin_amdgcn_wmma_f32_16x16x32_f16(false, a, false, b,
                                                     (short)0, acc, false, false);
    }
    const int mbase = mtile << 4;
#pragma unroll
    for (int r = 0; r < 8; ++r)                   // D: VGPR r -> M=r+8*half
        H[(size_t)(mbase + r + half * 8) * HDC + n] = acc[r];
}

// ===========================================================================
// GEMM2: G[NN,40] = Hrelu[NN,128] @ W2[128,40]  (40 padded to 48; pad cols
// zero-filled at staging time so the WMMA path stays unguarded)
// ===========================================================================
__global__ void gat_gemm2_wmma(const float* __restrict__ X,
                               const float* __restrict__ W,
                               float* __restrict__ G)
{
    __shared__ __align__(32) _Float16 bsh[4 * 32 * 16];
    const int tid = threadIdx.x;
    const int n0  = blockIdx.y << 4;

#pragma unroll
    for (int j = 0; j < 8; ++j) {
        int t      = tid * 8 + j;
        int e      = t & 15;
        int lane_s = (t >> 4) & 31;
        int kt     = t >> 9;
        int k      = kt * 32 + krel_of(e, lane_s >> 4);
        int nn     = n0 + (lane_s & 15);
        float wv   = (nn < OUTC) ? W[(size_t)k * OUTC + nn] : 0.0f;
        bsh[t] = (_Float16)wv;
    }
    __syncthreads();

    const int wave  = tid >> 5;
    const int lane  = tid & 31;
    const int mtile = blockIdx.x * 8 + wave;
    if (mtile >= (NN / 16)) return;
    const int half = lane >> 4;
    const int mrow = (mtile << 4) + (lane & 15);
    const int n    = n0 + (lane & 15);

    v8f acc = {};
#pragma unroll
    for (int kt = 0; kt < 4; ++kt) {
        v16h a = load_a_frag(X, mrow, kt * 32, half);
        v16h b = *(const v16h*)&bsh[kt * 512 + lane * 16];
        acc = __builtin_amdgcn_wmma_f32_16x16x32_f16(false, a, false, b,
                                                     (short)0, acc, false, false);
    }
    const int mbase = mtile << 4;
    if (n < OUTC) {
#pragma unroll
        for (int r = 0; r < 8; ++r)
            G[(size_t)(mbase + r + half * 8) * OUTC + n] = acc[r];
    }
}

// ===========================================================================
// per-node attention halves, layer 1: alpha[n,h] = sum_c h[n,h,c]*a[h,c]
// ===========================================================================
__global__ void gat_alpha1(const float* __restrict__ H,
                           const float* __restrict__ a_src,
                           const float* __restrict__ a_dst,
                           float* __restrict__ als, float* __restrict__ ald)
{
    int idx = blockIdx.x * blockDim.x + threadIdx.x;
    if (idx >= NN * HEADS) return;
    int node = idx >> 3, h = idx & 7;
    const float* row = H + (size_t)node * HDC + h * HID;
    float s = 0.f, d = 0.f;
#pragma unroll
    for (int c = 0; c < HID; ++c) {
        float v = row[c];
        s += v * a_src[h * HID + c];
        d += v * a_dst[h * HID + c];
    }
    als[idx] = s; ald[idx] = d;
}

// edge src/dst fetch (self loops appended at the end)
__device__ __forceinline__ void edge_sd(const int* __restrict__ ei, int e, int& src, int& dst)
{
    if (e < EE) { src = ei[e]; dst = ei[EE + e]; }
    else        { src = dst = e - EE; }
}

// layer1 pass1: segment max of leaky logits (8 heads)
__global__ void gat_max1(const int* __restrict__ ei, const float* __restrict__ als,
                         const float* __restrict__ ald, int* __restrict__ m1)
{
    int idx = blockIdx.x * blockDim.x + threadIdx.x;
    if (idx >= ETOT * HEADS) return;
    int e = idx >> 3, h = idx & 7, src, dst;
    edge_sd(ei, e, src, dst);
    float lg = lrelu(als[src * HEADS + h] + ald[dst * HEADS + h]);
    atomicMax(&m1[dst * HEADS + h], fenc(lg));
}

// layer1 pass2: denom = segment sum of exp(e - m)
__global__ void gat_sum1(const int* __restrict__ ei, const float* __restrict__ als,
                         const float* __restrict__ ald, const int* __restrict__ m1,
                         float* __restrict__ dn1)
{
    int idx = blockIdx.x * blockDim.x + threadIdx.x;
    if (idx >= ETOT * HEADS) return;
    int e = idx >> 3, h = idx & 7, src, dst;
    edge_sd(ei, e, src, dst);
    float lg = lrelu(als[src * HEADS + h] + ald[dst * HEADS + h]);
    atomicAdd(&dn1[dst * HEADS + h], expf(lg - fdec(m1[dst * HEADS + h])));
}

// layer1 pass3: out[dst,:] += alpha * h[src,:]   (wave per edge, 128 channels)
__global__ void gat_agg1(const int* __restrict__ ei, const float* __restrict__ als,
                         const float* __restrict__ ald, const int* __restrict__ m1,
                         const float* __restrict__ dn1, const float* __restrict__ H,
                         float* __restrict__ out)
{
    int wave = threadIdx.x >> 5, lane = threadIdx.x & 31;
    int e = blockIdx.x * 8 + wave;
    if (e >= ETOT) return;
    int src, dst;
    edge_sd(ei, e, src, dst);
    // kick the gathered feature row toward the caches while we do softmax math
    __builtin_prefetch(H + (size_t)src * HDC + lane * 4, 0, 0);
#pragma unroll
    for (int i = 0; i < 4; ++i) {
        int c = i * 32 + lane;
        int h = c >> 4;
        float lg = lrelu(als[src * HEADS + h] + ald[dst * HEADS + h]);
        float w  = expf(lg - fdec(m1[dst * HEADS + h])) / dn1[dst * HEADS + h];
        atomicAdd(&out[(size_t)dst * HDC + c], w * H[(size_t)src * HDC + c]);
    }
}

// bias + relu in place (bias applied post-aggregation per PyG GATConv)
__global__ void gat_bias_relu(float* __restrict__ A, const float* __restrict__ b)
{
    int idx = blockIdx.x * blockDim.x + threadIdx.x;
    if (idx >= NN * HDC) return;
    float v = A[idx] + b[idx & (HDC - 1)];
    A[idx] = v > 0.f ? v : 0.f;
}

// per-node attention halves, layer 2 (1 head, 40 channels)
__global__ void gat_alpha2(const float* __restrict__ G, const float* __restrict__ a_src,
                           const float* __restrict__ a_dst,
                           float* __restrict__ als, float* __restrict__ ald)
{
    int n = blockIdx.x * blockDim.x + threadIdx.x;
    if (n >= NN) return;
    const float* row = G + (size_t)n * OUTC;
    float s = 0.f, d = 0.f;
#pragma unroll
    for (int c = 0; c < OUTC; ++c) { float v = row[c]; s += v * a_src[c]; d += v * a_dst[c]; }
    als[n] = s; ald[n] = d;
}

__global__ void gat_max2(const int* __restrict__ ei, const float* __restrict__ als,
                         const float* __restrict__ ald, int* __restrict__ m2)
{
    int e = blockIdx.x * blockDim.x + threadIdx.x;
    if (e >= ETOT) return;
    int src, dst; edge_sd(ei, e, src, dst);
    atomicMax(&m2[dst], fenc(lrelu(als[src] + ald[dst])));
}

__global__ void gat_sum2(const int* __restrict__ ei, const float* __restrict__ als,
                         const float* __restrict__ ald, const int* __restrict__ m2,
                         float* __restrict__ dn2)
{
    int e = blockIdx.x * blockDim.x + threadIdx.x;
    if (e >= ETOT) return;
    int src, dst; edge_sd(ei, e, src, dst);
    float lg = lrelu(als[src] + ald[dst]);
    atomicAdd(&dn2[dst], expf(lg - fdec(m2[dst])));
}

// layer2 aggregate straight into d_out (wave per edge, 40 channels)
__global__ void gat_agg2(const int* __restrict__ ei, const float* __restrict__ als,
                         const float* __restrict__ ald, const int* __restrict__ m2,
                         const float* __restrict__ dn2, const float* __restrict__ G,
                         float* __restrict__ out)
{
    int wave = threadIdx.x >> 5, lane = threadIdx.x & 31;
    int e = blockIdx.x * 8 + wave;
    if (e >= ETOT) return;
    int src, dst; edge_sd(ei, e, src, dst);
    __builtin_prefetch(G + (size_t)src * OUTC + lane, 0, 0);
    float lg = lrelu(als[src] + ald[dst]);
    float w  = expf(lg - fdec(m2[dst])) / dn2[dst];
    int c = lane;
    if (c < OUTC) atomicAdd(&out[(size_t)dst * OUTC + c], w * G[(size_t)src * OUTC + c]);
    c = lane + 32;
    if (c < OUTC) atomicAdd(&out[(size_t)dst * OUTC + c], w * G[(size_t)src * OUTC + c]);
}

// final: row += b2, then log_softmax over 40 channels, in place
__global__ void gat_logsoftmax(float* __restrict__ out, const float* __restrict__ b2)
{
    int n = blockIdx.x * blockDim.x + threadIdx.x;
    if (n >= NN) return;
    float* row = out + (size_t)n * OUTC;
    float mx = -3.4e38f;
#pragma unroll
    for (int c = 0; c < OUTC; ++c) { float v = row[c] + b2[c]; mx = v > mx ? v : mx; }
    float s = 0.f;
#pragma unroll
    for (int c = 0; c < OUTC; ++c) s += expf(row[c] + b2[c] - mx);
    float ls = logf(s);
#pragma unroll
    for (int c = 0; c < OUTC; ++c) row[c] = row[c] + b2[c] - mx - ls;
}

// ===========================================================================
extern "C" void kernel_launch(void* const* d_in, const int* in_sizes, int n_in,
                              void* d_out, int out_size, void* d_ws, size_t ws_size,
                              hipStream_t stream)
{
    const float* x   = (const float*)d_in[0];
    const int*   ei  = (const int*)  d_in[1];
    const float* W1  = (const float*)d_in[2];
    const float* a1s = (const float*)d_in[3];
    const float* a1d = (const float*)d_in[4];
    const float* b1  = (const float*)d_in[5];
    const float* W2  = (const float*)d_in[6];
    const float* a2s = (const float*)d_in[7];
    const float* a2d = (const float*)d_in[8];
    const float* b2  = (const float*)d_in[9];
    float* out = (float*)d_out;

    // -------- workspace carving --------
    char* ws = (char*)d_ws;
    size_t off = 0;
    auto carve = [&](size_t elems) { void* p = ws + off; off += elems * sizeof(float); return p; };
    float* h1   = (float*)carve((size_t)NN * HDC);   // layer1 features
    float* agg1 = (float*)carve((size_t)NN * HDC);   // layer1 aggregate -> relu
    float* als1 = (float*)carve((size_t)NN * HEADS);
    float* ald1 = (float*)carve((size_t)NN * HEADS);
    int*   m1   = (int*)  carve((size_t)NN * HEADS);
    float* dn1  = (float*)carve((size_t)NN * HEADS);
    float* g2   = (float*)carve((size_t)NN * OUTC);  // layer2 features
    float* als2 = (float*)carve((size_t)NN);
    float* ald2 = (float*)carve((size_t)NN);
    int*   m2   = (int*)  carve((size_t)NN);
    float* dn2  = (float*)carve((size_t)NN);

    // -------- init accumulators (ws/d_out are poisoned by harness) --------
    hipMemsetAsync(agg1, 0,    (size_t)NN * HDC * 4, stream);
    hipMemsetAsync(dn1,  0,    (size_t)NN * HEADS * 4, stream);
    hipMemsetAsync(m1,   0x80, (size_t)NN * HEADS * 4, stream);  // decodes ~ -3.4e38
    hipMemsetAsync(dn2,  0,    (size_t)NN * 4, stream);
    hipMemsetAsync(m2,   0x80, (size_t)NN * 4, stream);
    hipMemsetAsync(out,  0,    (size_t)NN * OUTC * 4, stream);

    const int BT = 256;
    const int MT_GRID = (NN / 16 + 7) / 8;   // 391 wave-groups over 3125 M-tiles

    // ---- layer 1 ----
    gat_gemm1_wmma<<<dim3(MT_GRID, HDC / 16), BT, 0, stream>>>(x, W1, h1);
    gat_alpha1<<<(NN * HEADS + BT - 1) / BT, BT, 0, stream>>>(h1, a1s, a1d, als1, ald1);
    gat_max1<<<(ETOT * HEADS + BT - 1) / BT, BT, 0, stream>>>(ei, als1, ald1, m1);
    gat_sum1<<<(ETOT * HEADS + BT - 1) / BT, BT, 0, stream>>>(ei, als1, ald1, m1, dn1);
    gat_agg1<<<(ETOT + 7) / 8, BT, 0, stream>>>(ei, als1, ald1, m1, dn1, h1, agg1);
    gat_bias_relu<<<(NN * HDC + BT - 1) / BT, BT, 0, stream>>>(agg1, b1);

    // ---- layer 2 ----
    gat_gemm2_wmma<<<dim3(MT_GRID, 3), BT, 0, stream>>>(agg1, W2, g2);
    gat_alpha2<<<(NN + BT - 1) / BT, BT, 0, stream>>>(g2, a2s, a2d, als2, ald2);
    gat_max2<<<(ETOT + BT - 1) / BT, BT, 0, stream>>>(ei, als2, ald2, m2);
    gat_sum2<<<(ETOT + BT - 1) / BT, BT, 0, stream>>>(ei, als2, ald2, m2, dn2);
    gat_agg2<<<(ETOT + 7) / 8, BT, 0, stream>>>(ei, als2, ald2, m2, dn2, g2, out);
    gat_logsoftmax<<<(NN + BT - 1) / BT, BT, 0, stream>>>(out, b2);
}
